// ConformHopfieldBatchSameEnc_12816182411384
// MI455X (gfx1250) — matrix operand order repaired and checked by reading, not compiled, not run on hardware
//
#include <hip/hip_runtime.h>
#include <hip/hip_bf16.h>

typedef __attribute__((ext_vector_type(16))) _Float16 v16h;
typedef __attribute__((ext_vector_type(8)))  float    v8f;

#define WMMA(a,b,c) __builtin_amdgcn_wmma_f32_16x16x32_f16(false,(a),false,(b),(short)0,(c),false,false)

// ---------------- problem dimensions ----------------
constexpr int NB = 16, NM = 2, NS = 1024, NF = 32, ND = 64, NH = 4, NK = 20, NA = 10;
constexpr int HID = 100;          // hidden width (padded: K->128, N->112)
constexpr int HD  = 256;          // H*D
constexpr int STILES = NS / 16;   // 64 token tiles per (b)

// ---------------- output layout (floats) ----------------
constexpr size_t O_Y    = 2;
constexpr size_t O_LOW  = O_Y + (size_t)NB * NS * NM;                 // 32770
constexpr size_t O_HIGH = O_LOW + (size_t)NH * NB * NA * NS * NM;     // +1310720

// ---------------- workspace layout (bytes) ----------------
constexpr size_t OFF_W0  = 0;
constexpr size_t SZ_W0   = (size_t)NM * 7 * 1 * 512 * 2;
constexpr size_t OFF_W1  = OFF_W0 + SZ_W0;
constexpr size_t SZ_W12  = (size_t)NM * 7 * 4 * 512 * 2;
constexpr size_t OFF_W2  = OFF_W1 + SZ_W12;
constexpr size_t OFF_W3  = OFF_W2 + SZ_W12;
constexpr size_t SZ_W3   = (size_t)NM * 4 * 4 * 512 * 2;
constexpr size_t OFF_WQ  = OFF_W3 + SZ_W3;
constexpr size_t SZ_WQK  = (size_t)NM * 16 * 2 * 512 * 2;
constexpr size_t OFF_WK  = OFF_WQ + SZ_WQK;
constexpr size_t OFF_Q   = OFF_WK + SZ_WQK;                            // 292864 (256-aligned)
constexpr size_t SZ_QK   = (size_t)NB * NM * NH * STILES * 2 * 512 * 2; // 16 MB
constexpr size_t OFF_K   = OFF_Q + SZ_QK;
constexpr size_t OFF_ERR = OFF_K + SZ_QK;                              // B*M*H*S f32 (512 KB)

// ---------------- WMMA operand swizzle helpers (wave32, 16x16x32 f16) ----
// A (16x32): lane l holds row M=l&15; half j -> k = 2*(v&3)+h + 8*(l>>4) + 16*(v>>2), j=2v+h
__device__ __forceinline__ int a_k_of(int lane, int j) {
  int v = j >> 1, hb = j & 1;
  return ((v & 3) * 2 + hb) + ((lane >> 4) * 8) + ((v >> 2) * 16);
}
// inverse: position (half index in 512-half tile) of element (row, k_local)
__device__ __forceinline__ int a_pos(int row, int kl) {
  int vhi = kl >> 4;
  int r   = kl & 15;
  int lhi = r >> 3;
  int r2  = r & 7;
  int v   = vhi * 4 + (r2 >> 1);
  int j   = v * 2 + (r2 & 1);
  return (row + 16 * lhi) * 16 + j;
}
// B (32x16): lane l holds col N=l&15, K = (j>>3)*16 + (l>>4)*8 + (j&7)
__device__ __forceinline__ int b_pos(int kl, int col) {
  return (col + 16 * ((kl >> 3) & 1)) * 16 + ((kl >> 4) << 3) + (kl & 7);
}

union AView { _Float16 h[16]; v16h v; };

// ============================================================
// Kernel 1: convert f32 weights -> f16 B-operand tiles (zero padded)
// tile order: [(m*nt + ntile)*kc + chunk] * 512 halves, half p: lane=p>>4, idx=p&15
// ============================================================
__global__ __launch_bounds__(512) void prep_kernel(
    const float* __restrict__ W0, const float* __restrict__ W1,
    const float* __restrict__ W2, const float* __restrict__ W3,
    const float* __restrict__ Wq, const float* __restrict__ Wk,
    _Float16* __restrict__ w0p, _Float16* __restrict__ w1p,
    _Float16* __restrict__ w2p, _Float16* __restrict__ w3p,
    _Float16* __restrict__ wqp, _Float16* __restrict__ wkp)
{
  int t = blockIdx.x;
  const float* src; _Float16* dst; int Kd, Nd, kc, nt;
  if (t < 14)            { src = W0; dst = w0p; Kd = NF;  Nd = HID; kc = 1; nt = 7;  }
  else if ((t -= 14) < 56) { src = W1; dst = w1p; Kd = HID; Nd = HID; kc = 4; nt = 7;  }
  else if ((t -= 56) < 56) { src = W2; dst = w2p; Kd = HID; Nd = HID; kc = 4; nt = 7;  }
  else if ((t -= 56) < 32) { src = W3; dst = w3p; Kd = HID; Nd = ND;  kc = 4; nt = 4;  }
  else if ((t -= 32) < 64) { src = Wq; dst = wqp; Kd = ND;  Nd = HD;  kc = 2; nt = 16; }
  else            { t -= 64; src = Wk; dst = wkp; Kd = ND;  Nd = HD;  kc = 2; nt = 16; }
  const int chunk = t % kc;
  const int rest  = t / kc;
  const int ntile = rest % nt;
  const int mm    = rest / nt;
  const int p     = threadIdx.x;
  const int lane  = p >> 4, idx = p & 15;
  const int k = chunk * 32 + ((idx >> 3) << 4) + ((lane >> 4) << 3) + (idx & 7);
  const int n = ntile * 16 + (lane & 15);
  float v = 0.f;
  if (k < Kd && n < Nd) v = src[((size_t)mm * Kd + k) * Nd + n];
  dst[(size_t)t * 512 + p] = (_Float16)v;
}

// ============================================================
// Kernel 2: encoder (MLP + LN + Q/K projection), one wave per 16-token tile
// ============================================================
__global__ __launch_bounds__(128) void encoder_kernel(
    const float* __restrict__ X,
    const float* __restrict__ b0, const float* __restrict__ b1,
    const float* __restrict__ b2, const float* __restrict__ b3,
    const float* __restrict__ lnqg, const float* __restrict__ lnqb,
    const float* __restrict__ lnkg, const float* __restrict__ lnkb,
    const float* __restrict__ bq, const float* __restrict__ bk,
    const _Float16* __restrict__ w0p, const _Float16* __restrict__ w1p,
    const _Float16* __restrict__ w2p, const _Float16* __restrict__ w3p,
    const _Float16* __restrict__ wqp, const _Float16* __restrict__ wkp,
    _Float16* __restrict__ q_ws, _Float16* __restrict__ k_ws)
{
  __shared__ __align__(16) _Float16 s_act[4][2][2048]; // two A-swizzled act buffers / wave
  __shared__ __align__(16) float    s_enc[4][16 * ND];
  __shared__ float                  s_stats[4][32];

  const int lane  = threadIdx.x & 31;
  const int wv    = threadIdx.x >> 5;
  const int wid   = blockIdx.x * 4 + wv;      // 0..2047
  const int m     = wid >> 10;
  const int tile  = wid & 1023;
  const int b     = tile >> 6;
  const int stile = tile & 63;
  const int sbase = stile * 16;

  _Float16* act0 = s_act[wv][0];
  _Float16* act1 = s_act[wv][1];
  float*    enc   = s_enc[wv];
  float*    stats = s_stats[wv];

  { // zero both act buffers (guarantees K-padding rows 112..127 are 0)
    int4* z = (int4*)&s_act[wv][0][0];
    int4 z4 = {0, 0, 0, 0};
    for (int i = lane; i < 512; i += 32) z[i] = z4;
  }
  __syncthreads();

  // ---- layer 0: X(16x32) @ W0 -> act0 (cols 0..111, A-swizzled f16) ----
  AView aX;
  {
    const int t = sbase + (lane & 15);
    const float* xr = X + ((size_t)(b * NM + m) * NS + t) * NF;
#pragma unroll
    for (int j = 0; j < 16; ++j) aX.h[j] = (_Float16)xr[a_k_of(lane, j)];
  }
  for (int nt = 0; nt < 7; ++nt) {
    v8f c = {0.f, 0.f, 0.f, 0.f, 0.f, 0.f, 0.f, 0.f};
    v16h bt = *(const v16h*)(w0p + (size_t)(m * 7 + nt) * 512 + lane * 16);
    c = WMMA(aX.v, bt, c);
    const int col = nt * 16 + (lane & 15);
    const float bias = (col < HID) ? b0[m * HID + col] : 0.f;
#pragma unroll
    for (int v = 0; v < 8; ++v) {
      const int t = v + ((lane >> 4) << 3);
      float val = (col < HID) ? fmaxf(c[v] + bias, 0.f) : 0.f;
      act0[((col >> 5) * 512) + a_pos(t, col & 31)] = (_Float16)val;
    }
  }
  __syncthreads();

  auto dense_relu = [&](const _Float16* actin, _Float16* actout,
                        const _Float16* wp, const float* bb) {
    for (int nt = 0; nt < 7; ++nt) {
      v8f c = {0.f, 0.f, 0.f, 0.f, 0.f, 0.f, 0.f, 0.f};
      for (int ch = 0; ch < 4; ++ch) {
        v16h a  = *(const v16h*)(actin + ch * 512 + lane * 16);
        v16h bt = *(const v16h*)(wp + (size_t)((m * 7 + nt) * 4 + ch) * 512 + lane * 16);
        c = WMMA(a, bt, c);
      }
      const int col = nt * 16 + (lane & 15);
      const float bias = (col < HID) ? bb[m * HID + col] : 0.f;
#pragma unroll
      for (int v = 0; v < 8; ++v) {
        const int t = v + ((lane >> 4) << 3);
        float val = (col < HID) ? fmaxf(c[v] + bias, 0.f) : 0.f;
        actout[((col >> 5) * 512) + a_pos(t, col & 31)] = (_Float16)val;
      }
    }
    __syncthreads();
  };
  dense_relu(act0, act1, w1p, b1);   // layer 1
  dense_relu(act1, act0, w2p, b2);   // layer 2

  // ---- layer 3 -> enc (16x64, f32, row-major) ----
  for (int nt = 0; nt < 4; ++nt) {
    v8f c = {0.f, 0.f, 0.f, 0.f, 0.f, 0.f, 0.f, 0.f};
    for (int ch = 0; ch < 4; ++ch) {
      v16h a  = *(const v16h*)(act0 + ch * 512 + lane * 16);
      v16h bt = *(const v16h*)(w3p + (size_t)((m * 4 + nt) * 4 + ch) * 512 + lane * 16);
      c = WMMA(a, bt, c);
    }
    const int col = nt * 16 + (lane & 15);
    const float bias = b3[m * ND + col];
#pragma unroll
    for (int v = 0; v < 8; ++v) {
      const int t = v + ((lane >> 4) << 3);
      enc[t * ND + col] = c[v] + bias;
    }
  }
  __syncthreads();

  // ---- LN stats (shared by q/k paths) ----
  if (lane < 16) {
    float s0 = 0.f;
    for (int d = 0; d < ND; ++d) s0 += enc[lane * ND + d];
    const float mu = s0 * (1.f / ND);
    float v2 = 0.f;
    for (int d = 0; d < ND; ++d) { float dd = enc[lane * ND + d] - mu; v2 += dd * dd; }
    stats[lane * 2]     = mu;
    stats[lane * 2 + 1] = rsqrtf(v2 * (1.f / ND) + 1e-5f);
  }
  __syncthreads();

  // ---- q/k projection; scatter pre-swizzled operands to workspace ----
  auto project = [&](const float* g, const float* bb, const float* biasP,
                     const _Float16* wp, _Float16* dst, bool asA) {
    const int row = lane & 15;
    const float mu = stats[row * 2], rs = stats[row * 2 + 1];
    AView a0, a1;
#pragma unroll
    for (int j = 0; j < 16; ++j) {
      const int kl = a_k_of(lane, j);
      { int d = kl;      a0.h[j] = (_Float16)(((enc[row * ND + d] - mu) * rs) * g[m * ND + d] + bb[m * ND + d]); }
      { int d = kl + 32; a1.h[j] = (_Float16)(((enc[row * ND + d] - mu) * rs) * g[m * ND + d] + bb[m * ND + d]); }
    }
    for (int nt = 0; nt < 16; ++nt) {
      v8f c = {0.f, 0.f, 0.f, 0.f, 0.f, 0.f, 0.f, 0.f};
      v16h bt0 = *(const v16h*)(wp + (size_t)((m * 16 + nt) * 2 + 0) * 512 + lane * 16);
      c = WMMA(a0.v, bt0, c);
      v16h bt1 = *(const v16h*)(wp + (size_t)((m * 16 + nt) * 2 + 1) * 512 + lane * 16);
      c = WMMA(a1.v, bt1, c);
      const int col = nt * 16 + (lane & 15);
      const float bias = biasP[m * HD + col];
      const int hh = col >> 6, d = col & 63, ch = d >> 5, kl = d & 31;
      const size_t tb = ((((size_t)(b * NM + m) * NH + hh) * STILES + stile) * 2 + ch) * 512;
#pragma unroll
      for (int v = 0; v < 8; ++v) {
        const int t = v + ((lane >> 4) << 3);
        const int pos = asA ? a_pos(t, kl) : b_pos(kl, t);
        dst[tb + pos] = (_Float16)(c[v] + bias);
      }
    }
  };
  project(lnqg, lnqb, bq, wqp, q_ws, true);
  project(lnkg, lnkb, bk, wkp, k_ws, false);
}

// ============================================================
// Kernel 3: scores (WMMA) + top-20 + quantiles.  1 wave / (b,m,h,stile).
// softmax elided: monotone => identical top-k indices.
// ============================================================
__device__ __forceinline__ float quant20(const float* v, float q) {
  float pos = q * 19.0f;
  int i = (int)pos;              // pos >= 0
  float f = pos - (float)i;
  int i2 = (i + 1 < 20) ? i + 1 : 19;
  return v[i] + f * (v[i2] - v[i]);
}

__global__ __launch_bounds__(32) void attn_topk_kernel(
    const _Float16* __restrict__ q_ws, const _Float16* __restrict__ k_ws,
    const float* __restrict__ Y, float* __restrict__ out, float* __restrict__ errw)
{
  extern __shared__ float sc[];              // 16 x 1025 (padded stride vs 64 banks)
  constexpr int RSTR = 1025;
  const int lane = threadIdx.x;
  int id = blockIdx.x;
  const int stile = id & 63;
  const int h = (id >> 6) & 3;
  const int m = (id >> 8) & 1;
  const int b = id >> 9;

  const size_t qtile = (((size_t)(b * NM + m) * NH + h) * STILES + stile) * 2;
  v16h a0 = *(const v16h*)(q_ws + qtile * 512 + lane * 16);
  v16h a1 = *(const v16h*)(q_ws + (qtile + 1) * 512 + lane * 16);

  const size_t kbase = (((size_t)(b * NM + m) * NH + h) * STILES) * 2 * 512;
  for (int tt = 0; tt < STILES; ++tt) {
    const _Float16* kp = k_ws + kbase + (size_t)tt * 1024;
    if (tt + 1 < STILES) __builtin_prefetch(kp + 1024 + lane * 16, 0, 0);
    v16h bt0 = *(const v16h*)(kp + lane * 16);
    v16h bt1 = *(const v16h*)(kp + 512 + lane * 16);
    v8f c = {0.f, 0.f, 0.f, 0.f, 0.f, 0.f, 0.f, 0.f};
    c = WMMA(a0, bt0, c);
    c = WMMA(a1, bt1, c);
    const int col = lane & 15;
    const int rbase = (lane >> 4) << 3;
#pragma unroll
    for (int v = 0; v < 8; ++v)
      sc[(rbase + v) * RSTR + tt * 16 + col] = c[v] * 0.125f;   // beta = 1/sqrt(64)
  }
  __syncthreads();

  if (lane < 16) {
    // streaming top-20 (sorted ascending; strict > keeps lowest tied index)
    float tv[NK]; int ti[NK];
#pragma unroll
    for (int i = 0; i < NK; ++i) { tv[i] = -3.0e38f; ti[i] = 0; }
    const float* row = sc + lane * RSTR;
    for (int j = 0; j < NS; ++j) {
      float v = row[j];
      if (v > tv[0]) {
        int p = 0;
        while (p < NK - 1 && tv[p + 1] < v) ++p;
        for (int q2 = 0; q2 < p; ++q2) { tv[q2] = tv[q2 + 1]; ti[q2] = ti[q2 + 1]; }
        tv[p] = v; ti[p] = j;
      }
    }
    const float* yrow = Y + (size_t)(b * NM + m) * NS;
    float sel[NK];
    for (int i = 0; i < NK; ++i) sel[i] = yrow[ti[i]];
    for (int i = 1; i < NK; ++i) {            // insertion sort ascending
      float x = sel[i]; int j = i - 1;
      while (j >= 0 && sel[j] > x) { sel[j + 1] = sel[j]; --j; }
      sel[j + 1] = x;
    }

    const float AL[NA] = {0.05f, 0.06f, 0.08f, 0.10f, 0.12f, 0.14f, 0.15f, 0.17f, 0.19f, 0.20f};
    const int s = stile * 16 + lane;
    float sumq = 0.f;
#pragma unroll
    for (int a = 0; a < NA; ++a) {
      float ql = quant20(sel, AL[a] * 0.5f);
      float qh = quant20(sel, 1.0f - AL[a] * 0.5f);
      sumq += ql + qh;
      size_t idx = ((((size_t)h * NB + b) * NA + a) * NS + s) * NM + m;
      out[O_LOW + idx]  = ql;
      out[O_HIGH + idx] = qh;
    }
    const float ypred = sumq * (1.0f / (2 * NA));
    const float d0 = yrow[s] - ypred;
    errw[(((size_t)m * NB + b) * NH + h) * NS + s] = d0 * d0;
  }
}

// ============================================================
// Kernel 4: y_out transpose;  Kernel 5: deterministic scores reduction
// ============================================================
__global__ void yinit_kernel(const float* __restrict__ Y, float* __restrict__ out) {
  int i = blockIdx.x * blockDim.x + threadIdx.x;
  if (i < NB * NS * NM) {
    const int m = i % NM;
    const int s = (i / NM) % NS;
    const int b = i / (NM * NS);
    out[O_Y + i] = Y[((size_t)(b * NM + m)) * NS + s];
  }
}

__global__ __launch_bounds__(256) void reduce_kernel(const float* __restrict__ errw,
                                                     float* __restrict__ out) {
  __shared__ float red[256];
  const int m = blockIdx.x, t = threadIdx.x;
  const int n = NB * NH * NS;  // 65536
  float s = 0.f;
  for (int i = t; i < n; i += 256) s += errw[(size_t)m * n + i];
  red[t] = s;
  __syncthreads();
  for (int w = 128; w > 0; w >>= 1) { if (t < w) red[t] += red[t + w]; __syncthreads(); }
  if (t == 0) out[m] = red[0] / (float)n;
}

// ============================================================
extern "C" void kernel_launch(void* const* d_in, const int* in_sizes, int n_in,
                              void* d_out, int out_size, void* d_ws, size_t ws_size,
                              hipStream_t stream) {
  (void)in_sizes; (void)n_in; (void)out_size; (void)ws_size;
  const float* X    = (const float*)d_in[0];
  const float* Y    = (const float*)d_in[1];
  const float* W0   = (const float*)d_in[2];
  const float* b0   = (const float*)d_in[3];
  const float* W1   = (const float*)d_in[4];
  const float* b1   = (const float*)d_in[5];
  const float* W2   = (const float*)d_in[6];
  const float* b2   = (const float*)d_in[7];
  const float* W3   = (const float*)d_in[8];
  const float* b3   = (const float*)d_in[9];
  const float* lnqg = (const float*)d_in[10];
  const float* lnqb = (const float*)d_in[11];
  const float* lnkg = (const float*)d_in[12];
  const float* lnkb = (const float*)d_in[13];
  const float* Wq   = (const float*)d_in[14];
  const float* bq   = (const float*)d_in[15];
  const float* Wk   = (const float*)d_in[16];
  const float* bk   = (const float*)d_in[17];
  float* out = (float*)d_out;
  char*  ws  = (char*)d_ws;

  _Float16* w0p = (_Float16*)(ws + OFF_W0);
  _Float16* w1p = (_Float16*)(ws + OFF_W1);
  _Float16* w2p = (_Float16*)(ws + OFF_W2);
  _Float16* w3p = (_Float16*)(ws + OFF_W3);
  _Float16* wqp = (_Float16*)(ws + OFF_WQ);
  _Float16* wkp = (_Float16*)(ws + OFF_WK);
  _Float16* qws = (_Float16*)(ws + OFF_Q);
  _Float16* kws = (_Float16*)(ws + OFF_K);
  float*    errw = (float*)(ws + OFF_ERR);

  prep_kernel<<<286, 512, 0, stream>>>(W0, W1, W2, W3, Wq, Wk,
                                       w0p, w1p, w2p, w3p, wqp, wkp);
  encoder_kernel<<<512, 128, 0, stream>>>(X, b0, b1, b2, b3,
                                          lnqg, lnqb, lnkg, lnkb, bq, bk,
                                          w0p, w1p, w2p, w3p, wqp, wkp, qws, kws);
  yinit_kernel<<<(NB * NS * NM + 255) / 256, 256, 0, stream>>>(Y, out);
  attn_topk_kernel<<<NB * NM * NH * STILES, 32, 16 * 1025 * sizeof(float), stream>>>(
      qws, kws, Y, out, errw);
  reduce_kernel<<<NM, 256, 0, stream>>>(errw, out);
}
